// GATLinkPredictor_41034117546282
// MI455X (gfx1250) — matrix operand order repaired and checked by reading, not compiled, not run on hardware
//
#include <hip/hip_runtime.h>
#include <hip/hip_bf16.h>

// ---------------------------------------------------------------------------
// GAT link predictor for MI455X (gfx1250, wave32, WMMA).
// GEMMs: v_wmma_f32_16x16x32_f16, f32->f16 operand convert, f32 accumulate.
// Each wave computes a 16x32 output tile (2 accumulators, shared A fragment).
// A fragments loaded as float4 (global_load_b128); edge softmax via atomics.
// ---------------------------------------------------------------------------

typedef __attribute__((ext_vector_type(16))) _Float16 v16h;
typedef __attribute__((ext_vector_type(8)))  float    v8f;

static __device__ __forceinline__ _Float16 f2h(float f) { return (_Float16)f; }

static __device__ __forceinline__ v16h pack16(float4 a, float4 b,
                                              float4 c, float4 d) {
  v16h r;
  r[0]  = f2h(a.x); r[1]  = f2h(a.y); r[2]  = f2h(a.z); r[3]  = f2h(a.w);
  r[4]  = f2h(b.x); r[5]  = f2h(b.y); r[6]  = f2h(b.z); r[7]  = f2h(b.w);
  r[8]  = f2h(c.x); r[9]  = f2h(c.y); r[10] = f2h(c.z); r[11] = f2h(c.w);
  r[12] = f2h(d.x); r[13] = f2h(d.y); r[14] = f2h(d.z); r[15] = f2h(d.w);
  return r;
}

// float atomic-max via sign-aware integer atomics (correct for mixed signs)
static __device__ __forceinline__ void atomicMaxF(float* addr, float v) {
  if (v >= 0.0f) atomicMax((int*)addr, __float_as_int(v));
  else           atomicMin((unsigned int*)addr, __float_as_uint(v));
}

// ---------------------------------------------------------------------------
// WMMA GEMM: C[M,Ncols] = A[M,K] * B[K,Ncols], row-major f32, f16 operands.
// One wave -> 16x32 tile (two 16x16 WMMA accumulators). K % 32 == 0,
// Ncols % 32 == 0.
// ---------------------------------------------------------------------------
__global__ __launch_bounds__(256)
void gemm_wmma_f16(const float* __restrict__ A, const float* __restrict__ B,
                   float* __restrict__ C, int M, int Ncols, int K,
                   int Mtiles, int Ntiles2) {
  int wave = (int)((blockIdx.x * (unsigned)blockDim.x + threadIdx.x) >> 5);
  if (wave >= Mtiles * Ntiles2) return;           // wave-uniform exit
  int lane = threadIdx.x & 31;
  int mt = wave / Ntiles2, nt = wave % Ntiles2;
  int m0 = mt << 4, n0 = nt << 5;
  int row = lane & 15, hi = lane >> 4, col = lane & 15;
  int ar = m0 + row; if (ar >= M) ar = M - 1;     // clamp (no divergence)
  const float* Arow = A + (size_t)ar * K;
  v8f acc0 = {}, acc1 = {};
#pragma unroll
  for (int k0 = 0; k0 < 128; k0 += 32) {
    if (k0 >= K) break;
    // A 16x32 f16 fragment: per lane two contiguous 8-float blocks ->
    // K = k0+8*hi .. +7 (elems 0-7) and K = k0+16+8*hi .. +7 (elems 8-15)
    const float4* Alo = (const float4*)(Arow + k0 + 8 * hi);
    const float4* Ahi = (const float4*)(Arow + k0 + 16 + 8 * hi);
    v16h af = pack16(Alo[0], Alo[1], Ahi[0], Ahi[1]);
    // B 32x16 fragments: elems 0-7 K=8*hi..+7, elems 8-15 K=16+8*hi..+7
    const float* Bp0 = B + (size_t)(k0 + 8 * hi) * Ncols + n0 + col;
    const float* Bp1 = Bp0 + 16;
    v16h bf0, bf1;
#pragma unroll
    for (int e = 0; e < 16; ++e) {
      bf0[e] = f2h(Bp0[(size_t)(e < 8 ? e : e + 8) * Ncols]);
      bf1[e] = f2h(Bp1[(size_t)(e < 8 ? e : e + 8) * Ncols]);
    }
    acc0 = __builtin_amdgcn_wmma_f32_16x16x32_f16(false, af, false, bf0,
                                                  (short)0, acc0, false, false);
    acc1 = __builtin_amdgcn_wmma_f32_16x16x32_f16(false, af, false, bf1,
                                                  (short)0, acc1, false, false);
  }
  // C/D layout: VGPR r -> M = m0 + 8*hi + r, N = n0 + col (+16 for acc1)
  if (m0 + 16 <= M) {                              // fast path: full tile
#pragma unroll
    for (int r = 0; r < 8; ++r) {
      size_t off = (size_t)(m0 + 8 * hi + r) * Ncols + n0 + col;
      C[off]      = acc0[r];
      C[off + 16] = acc1[r];
    }
  } else {
#pragma unroll
    for (int r = 0; r < 8; ++r) {
      int m = m0 + 8 * hi + r;
      if (m < M) {
        size_t off = (size_t)m * Ncols + n0 + col;
        C[off]      = acc0[r];
        C[off + 16] = acc1[r];
      }
    }
  }
}

// attention coefficients: as[n,h] = dot(xl[n,h,:], a_src[h,:]) (same for ad)
__global__ __launch_bounds__(256)
void alpha_kernel(const float* __restrict__ xl, const float* __restrict__ a_src,
                  const float* __restrict__ a_dst, float* __restrict__ as_,
                  float* __restrict__ ad_, int N, int H, int C) {
  int t = blockIdx.x * blockDim.x + threadIdx.x;
  if (t >= N * H) return;
  int n = t / H, h = t % H;
  const float* xr = xl + (size_t)n * H * C + (size_t)h * C;
  float s0 = 0.f, s1 = 0.f;
  for (int c = 0; c < C; ++c) {
    float v = xr[c];
    s0 += v * a_src[h * C + c];
    s1 += v * a_dst[h * C + c];
  }
  as_[t] = s0; ad_[t] = s1;
}

__global__ __launch_bounds__(256)
void fill_u32(unsigned int* __restrict__ p, unsigned int v, long long n) {
  long long t = blockIdx.x * (long long)blockDim.x + threadIdx.x;
  if (t < n) p[t] = v;
}

static __device__ __forceinline__ void get_edge(const int* ei, int E,
                                                long long e, int& s, int& d) {
  if (e < (long long)E) { s = ei[e]; d = ei[(long long)E + e]; }
  else { s = d = (int)(e - E); }  // appended self loops
}

// pass 1: segment max of leaky-relu logits over destination nodes
__global__ __launch_bounds__(256)
void edge_max_kernel(const int* __restrict__ ei, int E, int N, int H,
                     const float* __restrict__ as_, const float* __restrict__ ad_,
                     float* __restrict__ m) {
  long long t = blockIdx.x * (long long)blockDim.x + threadIdx.x;
  long long total = (long long)(E + N) * H;
  if (t >= total) return;
  int h = (int)(t % H); long long e = t / H;
  int sv, dv; get_edge(ei, E, e, sv, dv);
  float v = as_[(size_t)sv * H + h] + ad_[(size_t)dv * H + h];
  v = (v >= 0.f) ? v : 0.2f * v;
  atomicMaxF(&m[(size_t)dv * H + h], v);
}

// pass 2: one wave per (edge, head): accumulate exp sums and weighted features
__global__ __launch_bounds__(256)
void edge_acc_kernel(const int* __restrict__ ei, int E, int N, int H, int C,
                     const float* __restrict__ as_, const float* __restrict__ ad_,
                     const float* __restrict__ m, const float* __restrict__ xl,
                     float* __restrict__ s, float* __restrict__ acc) {
  long long w = (blockIdx.x * (long long)blockDim.x + threadIdx.x) >> 5;
  int lane = threadIdx.x & 31;
  long long total = (long long)(E + N) * H;
  if (w >= total) return;
  int h = (int)(w % H); long long e = w / H;
  int sv, dv; get_edge(ei, E, e, sv, dv);
  float ev = as_[(size_t)sv * H + h] + ad_[(size_t)dv * H + h];
  ev = (ev >= 0.f) ? ev : 0.2f * ev;
  float ex = __expf(ev - m[(size_t)dv * H + h]);
  if (lane == 0) atomicAdd(&s[(size_t)dv * H + h], ex);
  const float* xr = xl + (size_t)sv * H * C + (size_t)h * C;
  float* ar = acc + (size_t)dv * H * C + (size_t)h * C;
  for (int c = lane; c < C; c += 32) atomicAdd(&ar[c], xr[c] * ex);
}

// out[n,h,c] = act( acc[n,h,c] / s[n,h] + b[h*C+c] ), act = ELU or identity
__global__ __launch_bounds__(256)
void finalize_kernel(const float* __restrict__ acc, const float* __restrict__ s,
                     const float* __restrict__ b, float* __restrict__ out,
                     int N, int H, int C, int do_elu) {
  long long t = blockIdx.x * (long long)blockDim.x + threadIdx.x;
  long long total = (long long)N * H * C;
  if (t >= total) return;
  int hc = (int)(t % (H * C));
  long long n = t / (H * C);
  int h = hc / C;
  float v = acc[t] / s[n * H + h] + b[hc];
  if (do_elu) v = (v > 0.f) ? v : (__expf(v) - 1.f);
  out[t] = v;
}

// Fused link-prediction MLP: one wave handles 16 pred edges.
// A rows = concat(z[row],z[col]) (K=128) gathered on the fly; B = lw1[128,64].
// Layer 2 (64->1) done in-register: cross-lane reduce over the 16-lane groups.
__global__ __launch_bounds__(256)
void mlp_wmma_kernel(const float* __restrict__ z, const int* __restrict__ pei,
                     int Ep, const float* __restrict__ lw1,
                     const float* __restrict__ lb1, const float* __restrict__ lw2,
                     const float* __restrict__ lb2, float* __restrict__ out) {
  int wave = (int)((blockIdx.x * (unsigned)blockDim.x + threadIdx.x) >> 5);
  int Mtiles = (Ep + 15) >> 4;
  if (wave >= Mtiles) return;                      // wave-uniform exit
  int lane = threadIdx.x & 31;
  int m0 = wave << 4;
  int row = lane & 15, hi = lane >> 4, col = lane & 15;
  int am = m0 + row; if (am >= Ep) am = Ep - 1;
  int ri = pei[am], ci = pei[(size_t)Ep + am];
  const float* zr = z + (size_t)ri * 64;
  const float* zc = z + (size_t)ci * 64;
  float rowdot[8];
#pragma unroll
  for (int r = 0; r < 8; ++r) rowdot[r] = 0.f;
  for (int nt = 0; nt < 4; ++nt) {
    int n0 = nt << 4;
    v8f acc = {};
#pragma unroll
    for (int kk = 0; kk < 4; ++kk) {
      int k0 = kk << 5;
      // two contiguous 8-float blocks; each block (8-aligned) lies entirely
      // inside one 64-float half of concat(z[row], z[col])
      int slo = k0 + 8 * hi, shi = k0 + 16 + 8 * hi;
      const float4* Alo = (const float4*)((slo < 64) ? (zr + slo) : (zc + slo - 64));
      const float4* Ahi = (const float4*)((shi < 64) ? (zr + shi) : (zc + shi - 64));
      v16h af = pack16(Alo[0], Alo[1], Ahi[0], Ahi[1]);
      const float* Bp = lw1 + (size_t)(k0 + 8 * hi) * 64 + n0 + col;
      v16h bf;
#pragma unroll
      for (int e = 0; e < 16; ++e) bf[e] = f2h(Bp[(size_t)(e < 8 ? e : e + 8) * 64]);
      acc = __builtin_amdgcn_wmma_f32_16x16x32_f16(false, af, false, bf,
                                                   (short)0, acc, false, false);
    }
    float w2  = lw2[n0 + col];
    float b1v = lb1[n0 + col];
#pragma unroll
    for (int r = 0; r < 8; ++r) {
      float hm = acc[r] + b1v;
      hm = (hm > 0.f) ? hm : 0.f;                 // ReLU
      float p = hm * w2;                          // partial of 64-dot
      p += __shfl_xor(p, 1, 32);
      p += __shfl_xor(p, 2, 32);
      p += __shfl_xor(p, 4, 32);
      p += __shfl_xor(p, 8, 32);                  // reduce 16-lane group
      rowdot[r] += p;
    }
  }
  float lb2v = lb2[0];
  if ((lane & 15) == 0) {
#pragma unroll
    for (int r = 0; r < 8; ++r) {
      int mm = m0 + 8 * hi + r;
      if (mm < Ep) {
        float v = rowdot[r] + lb2v;
        out[mm] = 1.f / (1.f + __expf(-v));       // sigmoid
      }
    }
  }
}

// ---------------------------------------------------------------------------
extern "C" void kernel_launch(void* const* d_in, const int* in_sizes, int n_in,
                              void* d_out, int out_size, void* d_ws, size_t ws_size,
                              hipStream_t stream) {
  const float* x      = (const float*)d_in[0];
  const int*   ei     = (const int*)  d_in[1];
  const int*   pei    = (const int*)  d_in[2];
  const float* W1     = (const float*)d_in[3];
  const float* a_src1 = (const float*)d_in[4];
  const float* a_dst1 = (const float*)d_in[5];
  const float* b1     = (const float*)d_in[6];
  const float* W2     = (const float*)d_in[7];
  const float* a_src2 = (const float*)d_in[8];
  const float* a_dst2 = (const float*)d_in[9];
  const float* b2     = (const float*)d_in[10];
  const float* lw1    = (const float*)d_in[11];
  const float* lb1    = (const float*)d_in[12];
  const float* lw2    = (const float*)d_in[13];
  const float* lb2    = (const float*)d_in[14];
  float* out = (float*)d_out;

  const int Fin = 128, H = 2, Hid = 64, Out = 64;
  const int HC1 = H * Hid;                 // 128
  const int N  = in_sizes[0] / Fin;        // 50000
  const int E  = in_sizes[1] / 2;          // 800000
  const int Ep = in_sizes[2] / 2;          // 200000
  const int Etot = E + N;                  // + self loops

  // workspace carve-out
  char* wp = (char*)d_ws;
  auto alloc = [&](size_t bytes) {
    char* p = wp; wp += (bytes + 255) & ~(size_t)255; return p;
  };
  float* xl1  = (float*)alloc((size_t)N * HC1 * 4);   // layer1 features; reused as h1
  float* acc1 = (float*)alloc((size_t)N * HC1 * 4);
  float* as1  = (float*)alloc((size_t)N * H * 4);
  float* ad1  = (float*)alloc((size_t)N * H * 4);
  float* m1   = (float*)alloc((size_t)N * H * 4);
  float* s1   = (float*)alloc((size_t)N * H * 4);
  float* xl2  = (float*)alloc((size_t)N * Out * 4);   // layer2 features; reused as z
  float* acc2 = (float*)alloc((size_t)N * Out * 4);
  float* as2  = (float*)alloc((size_t)N * 4);
  float* ad2  = (float*)alloc((size_t)N * 4);
  float* m2   = (float*)alloc((size_t)N * 4);
  float* s2   = (float*)alloc((size_t)N * 4);
  (void)ws_size; (void)n_in; (void)out_size;

  const int BLK = 256;
  auto nblk = [](long long t, int b) { return (unsigned)((t + b - 1) / b); };

  // ---- layer 1 ----
  {
    int Mtiles = (N + 15) / 16, Ntiles2 = HC1 / 32;
    gemm_wmma_f16<<<nblk((long long)Mtiles * Ntiles2 * 32, BLK), BLK, 0, stream>>>(
        x, W1, xl1, N, HC1, Fin, Mtiles, Ntiles2);
  }
  alpha_kernel<<<nblk((long long)N * H, BLK), BLK, 0, stream>>>(
      xl1, a_src1, a_dst1, as1, ad1, N, H, Hid);
  fill_u32<<<nblk((long long)N * H, BLK), BLK, 0, stream>>>(
      (unsigned int*)m1, 0xFF800000u, (long long)N * H);          // -inf
  fill_u32<<<nblk((long long)N * H, BLK), BLK, 0, stream>>>(
      (unsigned int*)s1, 0u, (long long)N * H);
  fill_u32<<<nblk((long long)N * HC1, BLK), BLK, 0, stream>>>(
      (unsigned int*)acc1, 0u, (long long)N * HC1);
  edge_max_kernel<<<nblk((long long)Etot * H, BLK), BLK, 0, stream>>>(
      ei, E, N, H, as1, ad1, m1);
  edge_acc_kernel<<<nblk((long long)Etot * H * 32, BLK), BLK, 0, stream>>>(
      ei, E, N, H, Hid, as1, ad1, m1, xl1, s1, acc1);
  // h1 (ELU) overwrites xl1 (dead after edge_acc)
  finalize_kernel<<<nblk((long long)N * HC1, BLK), BLK, 0, stream>>>(
      acc1, s1, b1, xl1, N, H, Hid, 1);

  // ---- layer 2 (H=1) ----
  {
    int Mtiles = (N + 15) / 16, Ntiles2 = Out / 32;
    gemm_wmma_f16<<<nblk((long long)Mtiles * Ntiles2 * 32, BLK), BLK, 0, stream>>>(
        xl1, W2, xl2, N, Out, HC1, Mtiles, Ntiles2);
  }
  alpha_kernel<<<nblk((long long)N, BLK), BLK, 0, stream>>>(
      xl2, a_src2, a_dst2, as2, ad2, N, 1, Out);
  fill_u32<<<nblk((long long)N, BLK), BLK, 0, stream>>>(
      (unsigned int*)m2, 0xFF800000u, (long long)N);
  fill_u32<<<nblk((long long)N, BLK), BLK, 0, stream>>>(
      (unsigned int*)s2, 0u, (long long)N);
  fill_u32<<<nblk((long long)N * Out, BLK), BLK, 0, stream>>>(
      (unsigned int*)acc2, 0u, (long long)N * Out);
  edge_max_kernel<<<nblk((long long)Etot, BLK), BLK, 0, stream>>>(
      ei, E, N, 1, as2, ad2, m2);
  edge_acc_kernel<<<nblk((long long)Etot * 32, BLK), BLK, 0, stream>>>(
      ei, E, N, 1, Out, as2, ad2, m2, xl2, s2, acc2);
  // z overwrites xl2 (dead after edge_acc); no activation, mean over 1 head
  finalize_kernel<<<nblk((long long)N * Out, BLK), BLK, 0, stream>>>(
      acc2, s2, b2, xl2, N, 1, Out, 0);

  // ---- fused link-prediction MLP + sigmoid ----
  {
    int Mtiles = (Ep + 15) / 16;
    mlp_wmma_kernel<<<nblk((long long)Mtiles * 32, BLK), BLK, 0, stream>>>(
        xl2, pei, Ep, lw1, lb1, lw2, lb2, out);
  }
}
